// Rotary_LoFTREncoderLayer_vis_61426622267548
// MI455X (gfx1250) — matrix operand order repaired and checked by reading, not compiled, not run on hardware
//
#include <hip/hip_runtime.h>
#include <hip/hip_bf16.h>
#include <math.h>

// ---------------- problem constants ----------------
#define Nb   8
#define Lq   4096
#define Ssrc 4096
#define Cdim 256
#define Hn   8
#define Dh   32
#define NL   (Nb * Lq)    // 32768 rows (x side)
#define NS   (Nb * Ssrc)  // 32768 rows (source side)

typedef __attribute__((ext_vector_type(16))) __bf16 bf16x16;
typedef __attribute__((ext_vector_type(8)))  float  f32x8;
typedef __attribute__((ext_vector_type(4)))  int    i32x4;

union FragBF { uint4 q[2]; bf16x16 v; };

__device__ __forceinline__ unsigned short f2bf(float f) {
  __bf16 h = (__bf16)f;
  return __builtin_bit_cast(unsigned short, h);
}
__device__ __forceinline__ float bf2f(unsigned short u) {
  return (float)__builtin_bit_cast(__bf16, u);
}
__device__ __forceinline__ float eluplus(float x) {   // elu(x)+1
  return x > 0.f ? x + 1.f : __expf(x);
}

// ---------------- CDNA5 async global->LDS staging ----------------
#if defined(__AMDGCN__) && __has_builtin(__builtin_amdgcn_global_load_async_to_lds_b128) && \
    __has_builtin(__builtin_amdgcn_s_wait_asynccnt)
#define ATHENA_ASYNC_LDS 1
#else
#define ATHENA_ASYNC_LDS 0
#endif

// copy 16B from global to LDS (async DMA when available)
__device__ __forceinline__ void copy16_g2lds(const unsigned short* g, unsigned short* l) {
#if ATHENA_ASYNC_LDS
  typedef __attribute__((address_space(1))) i32x4* gas1_p;
  typedef __attribute__((address_space(3))) i32x4* las3_p;
  gas1_p gp = (gas1_p)(uintptr_t)g;
  las3_p lp = (las3_p)(uintptr_t)l;   // low 32 bits = LDS offset
  __builtin_amdgcn_global_load_async_to_lds_b128(gp, lp, 0, 0);
#else
  *(uint4*)l = *(const uint4*)g;
#endif
}
__device__ __forceinline__ void async_wait0() {
#if ATHENA_ASYNC_LDS
  __builtin_amdgcn_s_wait_asynccnt(0);
#endif
}

// ---------------- utility kernels ----------------
__global__ __launch_bounds__(256) void zero_f32_kernel(float* p, int n) {
  int i = blockIdx.x * 256 + threadIdx.x;
  if (i < n) p[i] = 0.f;
}

__global__ __launch_bounds__(256) void cvt_f32_bf16_kernel(const float* __restrict__ in,
                                                           unsigned short* __restrict__ out, int n) {
  int i = blockIdx.x * 256 + threadIdx.x;
  if (i < n) out[i] = f2bf(in[i]);
}

// ---------------- generic bf16 WMMA GEMM (double-buffered, async staging) ----------------
// A: [M,K] bf16 row-major, B: [K,N] bf16 row-major.
// mode 0: C fp32.  mode 1: relu(C) -> bf16.
// Block tile 128x64, 8 waves in 4x2 grid, each wave 32x32 (2x2 WMMA 16x16x32).
__global__ __launch_bounds__(256) void gemm_bf16_kernel(
    const unsigned short* __restrict__ A, const unsigned short* __restrict__ B,
    float* __restrict__ Cf, unsigned short* __restrict__ Cb,
    int M, int N, int K, int mode) {
  __shared__ __align__(16) unsigned short As[2][128][40];  // [buf][m][k], row pad to 40
  __shared__ __align__(16) unsigned short Bt[2][64][40];   // [buf][n][k] (B transposed)

  const int tid  = threadIdx.x;
  const int lane = tid & 31;
  const int wave = tid >> 5;
  const int wm = wave >> 1, wn = wave & 1;         // 4 x 2 wave grid
  const int m0 = blockIdx.y * 128;
  const int n0 = blockIdx.x * 64;
  const int lm = lane & 15, hh = lane >> 4;        // lane row/half

  f32x8 acc[2][2] = {};

  auto stage = [&](int k0, int buf) {
    // A tile: 128x32 bf16 = 512 x 16B, async DMA straight into LDS
#pragma unroll
    for (int i = 0; i < 2; ++i) {
      int idx = i * 256 + tid;
      int r = idx >> 2, c4 = idx & 3;
      copy16_g2lds(A + (size_t)(m0 + r) * K + k0 + c4 * 8, &As[buf][r][c4 * 8]);
    }
    // B tile transposed: 32(k) x 64(n), register transpose -> LDS scatter
    int kk = tid >> 3, n8 = (tid & 7) * 8;
    uint4 v = *(const uint4*)(B + (size_t)(k0 + kk) * N + n0 + n8);
    const unsigned short* s = (const unsigned short*)&v;
#pragma unroll
    for (int j = 0; j < 8; ++j) Bt[buf][n8 + j][kk] = s[j];
  };

  // prologue: stage first tile
  stage(0, 0);
  async_wait0();
  __syncthreads();

  int buf = 0;
  for (int k0 = 0; k0 < K; k0 += 32) {
    const int nxt = k0 + 32;
    if (nxt < K) {
      stage(nxt, buf ^ 1);              // overlap next-tile DMA with this tile's WMMAs
      if (nxt + 32 < K) {               // prefetch two tiles ahead
        __builtin_prefetch(A + (size_t)(m0 + (tid >> 1)) * K + nxt + 32, 0, 1);
        __builtin_prefetch(B + (size_t)(nxt + 32 + (tid >> 3)) * N + n0, 0, 1);
      }
    }

    // ---- fragments (ISA 16-bit A 16x32 / B 32x16 layouts) ----
    FragBF a[2], b[2];
#pragma unroll
    for (int ti = 0; ti < 2; ++ti) {
      int m = wm * 32 + ti * 16 + lm;
      a[ti].q[0] = *(const uint4*)(&As[buf][m][hh * 8]);        // K = 8h .. 8h+7
      a[ti].q[1] = *(const uint4*)(&As[buf][m][16 + hh * 8]);   // K = 16+8h ..
    }
#pragma unroll
    for (int tj = 0; tj < 2; ++tj) {
      int nn = wn * 32 + tj * 16 + lm;
      b[tj].q[0] = *(const uint4*)(&Bt[buf][nn][hh * 8]);       // K = 8h .. 8h+7
      b[tj].q[1] = *(const uint4*)(&Bt[buf][nn][16 + hh * 8]);  // K = 16+8h ..
    }
#pragma unroll
    for (int ti = 0; ti < 2; ++ti)
#pragma unroll
      for (int tj = 0; tj < 2; ++tj)
        acc[ti][tj] = __builtin_amdgcn_wmma_f32_16x16x32_bf16(
            false, a[ti].v, false, b[tj].v, (short)0, acc[ti][tj], false, false);

    if (nxt < K) {
      async_wait0();      // this wave's DMA for next tile done
      __syncthreads();    // all waves done reading buf / writing buf^1
    }
    buf ^= 1;
  }

  // ---- epilogue: C/D layout m = 8*h + r, n = lane&15 ----
#pragma unroll
  for (int ti = 0; ti < 2; ++ti)
#pragma unroll
    for (int tj = 0; tj < 2; ++tj)
#pragma unroll
      for (int r = 0; r < 8; ++r) {
        int m = m0 + wm * 32 + ti * 16 + hh * 8 + r;
        int n = n0 + wn * 32 + tj * 16 + lm;
        float v = acc[ti][tj][r];
        if (mode == 0) {
          Cf[(size_t)m * N + n] = v;
        } else {
          v = v > 0.f ? v : 0.f;
          Cb[(size_t)m * N + n] = f2bf(v);
        }
      }
}

// ---------------- Ksum[n,c] = sum_s elu(key[n,s,c])+1 ----------------
__global__ __launch_bounds__(256) void ksum_kernel(const float* __restrict__ projk,
                                                   float* __restrict__ Ksum) {
  const int chunks = Ssrc / 64;
  int n = blockIdx.x / chunks;
  int s0 = (blockIdx.x % chunks) * 64;
  int c = threadIdx.x;
  float acc = 0.f;
  for (int ss = 0; ss < 64; ++ss)
    acc += eluplus(projk[((size_t)(n * Ssrc + s0 + ss)) * Cdim + c]);
  atomicAdd(&Ksum[n * Cdim + c], acc);
}

// ---------------- rotary + feature map (query side) ----------------
__global__ __launch_bounds__(256) void rotq_kernel(const float* __restrict__ proj,
                                                   const float* __restrict__ pe,
                                                   unsigned short* __restrict__ Qf,
                                                   unsigned short* __restrict__ Qpos) {
  int i = blockIdx.x * 256 + threadIdx.x;
  if (i >= NL * (Cdim / 2)) return;
  int row = i / (Cdim / 2);
  int c0 = (i - row * (Cdim / 2)) * 2;
  size_t b = (size_t)row * Cdim + c0;
  float q0 = proj[b], q1 = proj[b + 1];
  const float* p = pe + b * 2;   // [..., c, {cos,sin}]
  float c_0 = p[0], s_0 = p[1], c_1 = p[2], s_1 = p[3];
  Qpos[b]     = f2bf(q0 * c_0 - q1 * s_0);
  Qpos[b + 1] = f2bf(q1 * c_1 + q0 * s_1);
  Qf[b]       = f2bf(eluplus(q0));
  Qf[b + 1]   = f2bf(eluplus(q1));
}

// ---------------- rotary (key + value/S) ----------------
__global__ __launch_bounds__(256) void rotkv_kernel(const float* __restrict__ projk,
                                                    const float* __restrict__ projv,
                                                    const float* __restrict__ pe,
                                                    unsigned short* __restrict__ Kpos,
                                                    unsigned short* __restrict__ Vrot,
                                                    float invS) {
  int i = blockIdx.x * 256 + threadIdx.x;
  if (i >= NS * (Cdim / 2)) return;
  int row = i / (Cdim / 2);
  int c0 = (i - row * (Cdim / 2)) * 2;
  size_t b = (size_t)row * Cdim + c0;
  const float* p = pe + b * 2;
  float c_0 = p[0], s_0 = p[1], c_1 = p[2], s_1 = p[3];
  float k0 = projk[b], k1 = projk[b + 1];
  Kpos[b]     = f2bf(k0 * c_0 - k1 * s_0);
  Kpos[b + 1] = f2bf(k1 * c_1 + k0 * s_1);
  float v0 = projv[b], v1 = projv[b + 1];
  Vrot[b]     = f2bf((v0 * c_0 - v1 * s_0) * invS);
  Vrot[b + 1] = f2bf((v1 * c_1 + v0 * s_1) * invS);
}

// ---------------- KV[n,h] = Kpos[n,:,h]^T @ Vrot[n,:,h]  (32x32, K=S) ----------------
__global__ __launch_bounds__(256) void kv_kernel(const unsigned short* __restrict__ Kpos,
                                                 const unsigned short* __restrict__ Vrot,
                                                 unsigned short* __restrict__ KVb) {
  __shared__ __align__(16) unsigned short KpT[8][32][40]; // [wave][d][s]
  __shared__ __align__(16) unsigned short VT[8][32][40];  // [wave][d][s]
  __shared__ float red[32][33];

  const int tid = threadIdx.x, lane = tid & 31, wave = tid >> 5;
  const int n = blockIdx.x / Hn, h = blockIdx.x % Hn;
  const int lm = lane & 15, hh = lane >> 4;

  for (int i = tid; i < 32 * 33; i += 256) ((float*)red)[i] = 0.f;

  f32x8 acc[2][2] = {};
  const size_t base = ((size_t)n * Ssrc) * Cdim + h * Dh;

  for (int s0 = wave * 32; s0 < Ssrc; s0 += 256) {
    __syncthreads();
    // each lane loads one s-row (32 bf16) of both matrices, scatters transposed
    const unsigned short* kp = Kpos + base + (size_t)(s0 + lane) * Cdim;
    const unsigned short* vp = Vrot + base + (size_t)(s0 + lane) * Cdim;
#pragma unroll
    for (int q4 = 0; q4 < 4; ++q4) {
      uint4 kv4 = *(const uint4*)(kp + q4 * 8);
      uint4 vv4 = *(const uint4*)(vp + q4 * 8);
      const unsigned short* ks = (const unsigned short*)&kv4;
      const unsigned short* vs = (const unsigned short*)&vv4;
#pragma unroll
      for (int j = 0; j < 8; ++j) {
        KpT[wave][q4 * 8 + j][lane] = ks[j];
        VT[wave][q4 * 8 + j][lane] = vs[j];
      }
    }
    __syncthreads();

    FragBF a[2], b[2];
#pragma unroll
    for (int ti = 0; ti < 2; ++ti) {
      int m = ti * 16 + lm;                         // A[m][k] = Kpos[s=k][d=m]
      a[ti].q[0] = *(const uint4*)(&KpT[wave][m][hh * 8]);
      a[ti].q[1] = *(const uint4*)(&KpT[wave][m][16 + hh * 8]);
    }
#pragma unroll
    for (int tj = 0; tj < 2; ++tj) {
      int nn = tj * 16 + lm;                        // B[k][nn] = Vrot[s=k][d=nn]
      b[tj].q[0] = *(const uint4*)(&VT[wave][nn][hh * 8]);
      b[tj].q[1] = *(const uint4*)(&VT[wave][nn][16 + hh * 8]);
    }
#pragma unroll
    for (int ti = 0; ti < 2; ++ti)
#pragma unroll
      for (int tj = 0; tj < 2; ++tj)
        acc[ti][tj] = __builtin_amdgcn_wmma_f32_16x16x32_bf16(
            false, a[ti].v, false, b[tj].v, (short)0, acc[ti][tj], false, false);
  }

  __syncthreads();
#pragma unroll
  for (int ti = 0; ti < 2; ++ti)
#pragma unroll
    for (int tj = 0; tj < 2; ++tj)
#pragma unroll
      for (int r = 0; r < 8; ++r) {
        int m = ti * 16 + hh * 8 + r;
        int nn = tj * 16 + lm;
        atomicAdd(&red[m][nn], acc[ti][tj][r]);
      }
  __syncthreads();
  for (int i = tid; i < 1024; i += 256) {
    int m = i >> 5, nn = i & 31;
    KVb[(((size_t)n * Hn + h) * 32 + m) * 32 + nn] = f2bf(red[m][nn]);
  }
}

// ---------------- Z[n,l,h] = 1 / (Qf . Ksum + eps) ----------------
__global__ __launch_bounds__(256) void z_kernel(const unsigned short* __restrict__ Qf,
                                                const float* __restrict__ Ksum,
                                                float* __restrict__ Z) {
  int i = blockIdx.x * 256 + threadIdx.x;
  if (i >= Nb * Lq * Hn) return;
  int h = i % Hn;
  int row = i / Hn;         // n*L + l
  int n = row / Lq;
  const unsigned short* q = Qf + (size_t)row * Cdim + h * Dh;
  const float* ks = Ksum + n * Cdim + h * Dh;
  float d = 0.f;
#pragma unroll
  for (int j = 0; j < Dh; ++j) d += bf2f(q[j]) * ks[j];
  Z[i] = 1.f / (d + 1e-6f);
}

// ---------------- queried = (Qpos @ KV) * Z * S -> bf16 ----------------
__global__ __launch_bounds__(32) void queried_kernel(const unsigned short* __restrict__ Qpos,
                                                     const unsigned short* __restrict__ KVb,
                                                     const float* __restrict__ Z,
                                                     unsigned short* __restrict__ outb,
                                                     float Sf) {
  __shared__ __align__(16) unsigned short Qs[32][40];   // [m][k]
  __shared__ __align__(16) unsigned short KVT[32][40];  // [nn][k]
  const int lane = threadIdx.x, lm = lane & 15, hh = lane >> 4;
  const int nt = Lq / 32;
  const int lt = blockIdx.x % nt;
  const int h = (blockIdx.x / nt) % Hn;
  const int n = blockIdx.x / (nt * Hn);
  const int l0 = lt * 32;

  // Q tile: async DMA straight into LDS (row-major, one row per lane)
  const unsigned short* qp = Qpos + ((size_t)n * Lq + l0 + lane) * Cdim + h * Dh;
#pragma unroll
  for (int q4 = 0; q4 < 4; ++q4) copy16_g2lds(qp + q4 * 8, &Qs[lane][q4 * 8]);

  // KV tile: register transpose
  const unsigned short* kv = KVb + ((size_t)n * Hn + h) * 1024 + lane * 32;
#pragma unroll
  for (int q4 = 0; q4 < 4; ++q4) {
    uint4 v = *(const uint4*)(kv + q4 * 8);
    const unsigned short* s = (const unsigned short*)&v;
#pragma unroll
    for (int j = 0; j < 8; ++j) KVT[q4 * 8 + j][lane] = s[j];
  }
  async_wait0();
  __syncthreads();

  f32x8 acc[2][2] = {};
  FragBF a[2], b[2];
#pragma unroll
  for (int ti = 0; ti < 2; ++ti) {
    int m = ti * 16 + lm;
    a[ti].q[0] = *(const uint4*)(&Qs[m][hh * 8]);
    a[ti].q[1] = *(const uint4*)(&Qs[m][16 + hh * 8]);
  }
#pragma unroll
  for (int tj = 0; tj < 2; ++tj) {
    int nn = tj * 16 + lm;
    b[tj].q[0] = *(const uint4*)(&KVT[nn][hh * 8]);
    b[tj].q[1] = *(const uint4*)(&KVT[nn][16 + hh * 8]);
  }
#pragma unroll
  for (int ti = 0; ti < 2; ++ti)
#pragma unroll
    for (int tj = 0; tj < 2; ++tj)
      acc[ti][tj] = __builtin_amdgcn_wmma_f32_16x16x32_bf16(
          false, a[ti].v, false, b[tj].v, (short)0, acc[ti][tj], false, false);

#pragma unroll
  for (int ti = 0; ti < 2; ++ti)
#pragma unroll
    for (int tj = 0; tj < 2; ++tj)
#pragma unroll
      for (int r = 0; r < 8; ++r) {
        int m = ti * 16 + hh * 8 + r;
        int l = l0 + m;
        int nn = tj * 16 + lm;
        float z = Z[((size_t)n * Lq + l) * Hn + h];
        outb[((size_t)n * Lq + l) * Cdim + h * Dh + nn] = f2bf(acc[ti][tj][r] * z * Sf);
      }
}

// ---------------- LayerNorm(256) + build concat [x | ln(msg)] (bf16) ----------------
__global__ __launch_bounds__(256) void ln1_cat_kernel(const float* __restrict__ msg,
                                                      const float* __restrict__ x,
                                                      const float* __restrict__ g,
                                                      const float* __restrict__ bb,
                                                      unsigned short* __restrict__ cat) {
  __shared__ float sred[256];
  const int row = blockIdx.x, tid = threadIdx.x;
  float v = msg[(size_t)row * Cdim + tid];
  sred[tid] = v; __syncthreads();
  for (int s = 128; s; s >>= 1) { if (tid < s) sred[tid] += sred[tid + s]; __syncthreads(); }
  float mean = sred[0] * (1.f / Cdim);
  __syncthreads();
  float d = v - mean;
  sred[tid] = d * d; __syncthreads();
  for (int s = 128; s; s >>= 1) { if (tid < s) sred[tid] += sred[tid + s]; __syncthreads(); }
  float var = sred[0] * (1.f / Cdim);
  float ln = d * rsqrtf(var + 1e-5f) * g[tid] + bb[tid];
  cat[(size_t)row * (2 * Cdim) + tid]        = f2bf(x[(size_t)row * Cdim + tid]);
  cat[(size_t)row * (2 * Cdim) + Cdim + tid] = f2bf(ln);
}

// ---------------- LayerNorm(256) + residual -> fp32 out ----------------
__global__ __launch_bounds__(256) void ln2_res_kernel(const float* __restrict__ m2,
                                                      const float* __restrict__ x,
                                                      const float* __restrict__ g,
                                                      const float* __restrict__ bb,
                                                      float* __restrict__ out) {
  __shared__ float sred[256];
  const int row = blockIdx.x, tid = threadIdx.x;
  float v = m2[(size_t)row * Cdim + tid];
  sred[tid] = v; __syncthreads();
  for (int s = 128; s; s >>= 1) { if (tid < s) sred[tid] += sred[tid + s]; __syncthreads(); }
  float mean = sred[0] * (1.f / Cdim);
  __syncthreads();
  float d = v - mean;
  sred[tid] = d * d; __syncthreads();
  for (int s = 128; s; s >>= 1) { if (tid < s) sred[tid] += sred[tid + s]; __syncthreads(); }
  float var = sred[0] * (1.f / Cdim);
  float ln = d * rsqrtf(var + 1e-5f) * g[tid] + bb[tid];
  out[(size_t)row * Cdim + tid] = x[(size_t)row * Cdim + tid] + ln;
}

// ---------------- launcher ----------------
extern "C" void kernel_launch(void* const* d_in, const int* in_sizes, int n_in,
                              void* d_out, int out_size, void* d_ws, size_t ws_size,
                              hipStream_t stream) {
  const float* x    = (const float*)d_in[0];
  const float* src  = (const float*)d_in[1];
  const float* x_pe = (const float*)d_in[2];
  const float* s_pe = (const float*)d_in[3];
  const float* Wq   = (const float*)d_in[4];
  const float* Wk   = (const float*)d_in[5];
  const float* Wv   = (const float*)d_in[6];
  const float* Wm   = (const float*)d_in[7];
  const float* W1   = (const float*)d_in[8];
  const float* W2   = (const float*)d_in[9];
  const float* ln1g = (const float*)d_in[10];
  const float* ln1b = (const float*)d_in[11];
  const float* ln2g = (const float*)d_in[12];
  const float* ln2b = (const float*)d_in[13];
  float* out = (float*)d_out;

  char* ws = (char*)d_ws;
  size_t off = 0;
  auto alloc = [&](size_t bytes) -> char* {
    char* p = ws + off;
    off += (bytes + 255) & ~(size_t)255;
    return p;
  };

  unsigned short* Wq_b = (unsigned short*)alloc((size_t)Cdim * Cdim * 2);
  unsigned short* Wk_b = (unsigned short*)alloc((size_t)Cdim * Cdim * 2);
  unsigned short* Wv_b = (unsigned short*)alloc((size_t)Cdim * Cdim * 2);
  unsigned short* Wm_b = (unsigned short*)alloc((size_t)Cdim * Cdim * 2);
  unsigned short* W1_b = (unsigned short*)alloc((size_t)512 * 512 * 2);
  unsigned short* W2_b = (unsigned short*)alloc((size_t)512 * 256 * 2);
  unsigned short* xb   = (unsigned short*)alloc((size_t)NL * Cdim * 2);   // adjacent to sb
  unsigned short* sb   = (unsigned short*)alloc((size_t)NS * Cdim * 2);
  float* projq = (float*)alloc((size_t)NL * Cdim * 4);
  float* projk = (float*)alloc((size_t)NS * Cdim * 4);
  float* projv = (float*)alloc((size_t)NS * Cdim * 4);
  unsigned short* Qf_b   = (unsigned short*)alloc((size_t)NL * Cdim * 2);
  unsigned short* Qpos_b = (unsigned short*)alloc((size_t)NL * Cdim * 2);
  unsigned short* Kpos_b = (unsigned short*)alloc((size_t)NS * Cdim * 2);
  unsigned short* Vrot_b = (unsigned short*)alloc((size_t)NS * Cdim * 2);
  float* Ksum = (float*)alloc((size_t)Nb * Cdim * 4);
  unsigned short* KVb = (unsigned short*)alloc((size_t)Nb * Hn * 32 * 32 * 2);
  float* Zbuf = (float*)alloc((size_t)Nb * Lq * Hn * 4);
  unsigned short* queried_b = (unsigned short*)alloc((size_t)NL * Cdim * 2);

  // safe aliases (regions dead by the time they're reused)
  float* msg = projq;                              // after rotq consumes projq
  unsigned short* cat_b = xb;                      // spans xb+sb (both dead after GEMMs)
  unsigned short* h_b = (unsigned short*)projv;    // dead after rotkv
  float* m2 = projk;                               // dead after rotkv/ksum

  // 1) bf16 conversions
  cvt_f32_bf16_kernel<<<(Cdim * Cdim + 255) / 256, 256, 0, stream>>>(Wq, Wq_b, Cdim * Cdim);
  cvt_f32_bf16_kernel<<<(Cdim * Cdim + 255) / 256, 256, 0, stream>>>(Wk, Wk_b, Cdim * Cdim);
  cvt_f32_bf16_kernel<<<(Cdim * Cdim + 255) / 256, 256, 0, stream>>>(Wv, Wv_b, Cdim * Cdim);
  cvt_f32_bf16_kernel<<<(Cdim * Cdim + 255) / 256, 256, 0, stream>>>(Wm, Wm_b, Cdim * Cdim);
  cvt_f32_bf16_kernel<<<(512 * 512 + 255) / 256, 256, 0, stream>>>(W1, W1_b, 512 * 512);
  cvt_f32_bf16_kernel<<<(512 * 256 + 255) / 256, 256, 0, stream>>>(W2, W2_b, 512 * 256);
  cvt_f32_bf16_kernel<<<(NL * Cdim + 255) / 256, 256, 0, stream>>>(x, xb, NL * Cdim);
  cvt_f32_bf16_kernel<<<(NS * Cdim + 255) / 256, 256, 0, stream>>>(src, sb, NS * Cdim);

  // 2) projections (WMMA GEMMs)
  dim3 gProj(Cdim / 64, NL / 128);
  gemm_bf16_kernel<<<gProj, 256, 0, stream>>>(xb, Wq_b, projq, nullptr, NL, Cdim, Cdim, 0);
  gemm_bf16_kernel<<<gProj, 256, 0, stream>>>(sb, Wk_b, projk, nullptr, NS, Cdim, Cdim, 0);
  gemm_bf16_kernel<<<gProj, 256, 0, stream>>>(sb, Wv_b, projv, nullptr, NS, Cdim, Cdim, 0);

  // 3) Ksum = sum_s elu(key)+1
  zero_f32_kernel<<<(Nb * Cdim + 255) / 256, 256, 0, stream>>>(Ksum, Nb * Cdim);
  ksum_kernel<<<Nb * (Ssrc / 64), 256, 0, stream>>>(projk, Ksum);

  // 4) rotary + feature maps
  rotq_kernel<<<(NL * (Cdim / 2) + 255) / 256, 256, 0, stream>>>(projq, x_pe, Qf_b, Qpos_b);
  rotkv_kernel<<<(NS * (Cdim / 2) + 255) / 256, 256, 0, stream>>>(projk, projv, s_pe, Kpos_b,
                                                                  Vrot_b, 1.0f / (float)Ssrc);

  // 5) KV = Kpos^T @ (V/S) per (n,h)  (WMMA, split-S + LDS reduce)
  kv_kernel<<<Nb * Hn, 256, 0, stream>>>(Kpos_b, Vrot_b, KVb);

  // 6) Z normalizer
  z_kernel<<<(Nb * Lq * Hn + 255) / 256, 256, 0, stream>>>(Qf_b, Ksum, Zbuf);

  // 7) queried = (Qpos @ KV) * Z * S  (WMMA)
  queried_kernel<<<Nb * Hn * (Lq / 32), 32, 0, stream>>>(Qpos_b, KVb, Zbuf, queried_b,
                                                         (float)Ssrc);

  // 8) message = queried @ Wm ; LN1 ; concat
  gemm_bf16_kernel<<<gProj, 256, 0, stream>>>(queried_b, Wm_b, msg, nullptr, NL, Cdim, Cdim, 0);
  ln1_cat_kernel<<<NL, 256, 0, stream>>>(msg, x, ln1g, ln1b, cat_b);

  // 9) FFN: relu(cat @ W1) @ W2
  dim3 gW1(512 / 64, NL / 128);
  gemm_bf16_kernel<<<gW1, 256, 0, stream>>>(cat_b, W1_b, nullptr, h_b, NL, 512, 512, 1);
  gemm_bf16_kernel<<<gProj, 256, 0, stream>>>(h_b, W2_b, m2, nullptr, NL, Cdim, 512, 0);

  // 10) LN2 + residual
  ln2_res_kernel<<<NL, 256, 0, stream>>>(m2, x, ln2g, ln2b, out);
}
